// ConformerEncoder_13314398617995
// MI455X (gfx1250) — matrix operand order repaired and checked by reading, not compiled, not run on hardware
//
#include <hip/hip_runtime.h>
#include <math.h>

// ---------------- model constants ----------------
constexpr int TSEQ  = 1023;   // tokens after 2x conv subsampling
constexpr int DM    = 512;    // model dim
constexpr int DFF   = 2048;   // 4*D
constexpr int NHEAD = 8;
constexpr int DKH   = 64;
constexpr int PLEN  = 2045;   // 2*T-1
constexpr int NL    = 12;
constexpr int DCONV = 1024;   // 2*D
constexpr int KC    = 33;
constexpr float EPS = 1e-5f;

// ---------------- WMMA types ----------------
typedef __attribute__((ext_vector_type(16))) __bf16 v16bf;
typedef __attribute__((ext_vector_type(8)))  float  v8f;
typedef __attribute__((ext_vector_type(8)))  unsigned short u16x8;
typedef __attribute__((ext_vector_type(4)))  unsigned short u16x4;

union FragU { v16bf v; u16x8 h[2]; };

__device__ __forceinline__ unsigned short f2bf(float f) {
  union { float f; unsigned u; } c; c.f = f;
  unsigned r = c.u + 0x7FFFu + ((c.u >> 16) & 1u);   // round-to-nearest-even
  return (unsigned short)(r >> 16);
}

constexpr int LDSS = 40;  // ushort stride: keeps 16B alignment, spreads LDS banks

// ---------------------------------------------------------------------------
// Generic batched GEMM: C[z] = A[z](MxK) * B[z](KxN) (+bias), bf16 WMMA, f32 acc
// flags bit0: B is (NxK) row-major (transposed-B access)
// flags bit1: add bias[col]
// ---------------------------------------------------------------------------
__global__ void __launch_bounds__(128)
gemm_bf16_kernel(const float* __restrict__ A, const float* __restrict__ B,
                 const float* __restrict__ bias, float* __restrict__ C,
                 int M, int N, int K, int lda, int ldb, int ldc,
                 long long sA, long long sB, long long sC, int flags)
{
  __shared__ __align__(16) unsigned short As[128 * LDSS];
  __shared__ __align__(16) unsigned short Bs[128 * LDSS];
  const int z = blockIdx.z;
  A += z * sA;  B += z * sB;  C += z * sC;
  const int mBase = blockIdx.y * 128;
  const int nBase = blockIdx.x * 128;
  const int tid  = threadIdx.x;
  const int lane = tid & 31;
  const int wav  = tid >> 5;
  const int l    = lane & 15;
  const int hi   = lane >> 4;
  const int wR   = (wav >> 1) * 64;   // wave M offset within block tile
  const int wC   = (wav & 1)  * 64;   // wave N offset within block tile
  const bool trB = (flags & 1) != 0;

  // uniform fast-path qualifiers (16B vector loads legal?)
  const bool fullM = (mBase + 128 <= M);
  const bool fullN = (nBase + 128 <= N);
  const bool vecA = ((lda & 3) == 0) && ((((unsigned long long)(const void*)A) & 15ull) == 0);
  const bool vecB = ((ldb & 3) == 0) && ((((unsigned long long)(const void*)B) & 15ull) == 0);

  v8f acc[4][4];
#pragma unroll
  for (int i = 0; i < 4; ++i)
#pragma unroll
    for (int j = 0; j < 4; ++j)
      acc[i][j] = v8f{0.f,0.f,0.f,0.f,0.f,0.f,0.f,0.f};

  const int nk = (K + 31) >> 5;
  for (int kb = 0; kb < nk; ++kb) {
    const int k0 = kb << 5;
    const bool fullK = (k0 + 32 <= K);
    __syncthreads();

    // ---- stage A tile (128x32) as bf16, row-major ----
    if (fullM & fullK & vecA) {
      // 1024 float4 loads total: each thread 8 x b128 global loads, b64 LDS stores
#pragma unroll
      for (int i = 0; i < 8; ++i) {
        int idx4 = i * 128 + tid;          // 0..1023
        int m    = idx4 >> 3;
        int kk   = (idx4 & 7) << 2;
        float4 v = *(const float4*)(A + (long long)(mBase + m) * lda + k0 + kk);
        u16x4 h; h[0] = f2bf(v.x); h[1] = f2bf(v.y); h[2] = f2bf(v.z); h[3] = f2bf(v.w);
        *(u16x4*)&As[m * LDSS + kk] = h;
      }
    } else {
#pragma unroll
      for (int i = 0; i < 32; ++i) {
        int e = i * 128 + tid;
        int m = e >> 5, kk = e & 31;
        int gm = mBase + m, gk = k0 + kk;
        float v = (gm < M && gk < K) ? A[(long long)gm * lda + gk] : 0.f;
        As[m * LDSS + kk] = f2bf(v);
      }
    }

    // ---- stage B tile transposed: Bs[n][k] ----
    if (!trB) {
      if (fullN & fullK & vecB) {
#pragma unroll
        for (int i = 0; i < 8; ++i) {
          int idx4 = i * 128 + tid;
          int kk   = idx4 >> 5;            // 32 float4 per 128-wide row
          int n0   = (idx4 & 31) << 2;
          float4 v = *(const float4*)(B + (long long)(k0 + kk) * ldb + nBase + n0);
          Bs[(n0 + 0) * LDSS + kk] = f2bf(v.x);
          Bs[(n0 + 1) * LDSS + kk] = f2bf(v.y);
          Bs[(n0 + 2) * LDSS + kk] = f2bf(v.z);
          Bs[(n0 + 3) * LDSS + kk] = f2bf(v.w);
        }
      } else {
#pragma unroll
        for (int i = 0; i < 32; ++i) {
          int e = i * 128 + tid;
          int n = e & 127, kk = e >> 7;
          int gn = nBase + n, gk = k0 + kk;
          float v = (gn < N && gk < K) ? B[(long long)gk * ldb + gn] : 0.f;
          Bs[n * LDSS + kk] = f2bf(v);
        }
      }
    } else {
      if (fullN & fullK & vecB) {
#pragma unroll
        for (int i = 0; i < 8; ++i) {
          int idx4 = i * 128 + tid;
          int n    = idx4 >> 3;
          int kk   = (idx4 & 7) << 2;
          float4 v = *(const float4*)(B + (long long)(nBase + n) * ldb + k0 + kk);
          u16x4 h; h[0] = f2bf(v.x); h[1] = f2bf(v.y); h[2] = f2bf(v.z); h[3] = f2bf(v.w);
          *(u16x4*)&Bs[n * LDSS + kk] = h;
        }
      } else {
#pragma unroll
        for (int i = 0; i < 32; ++i) {
          int e = i * 128 + tid;
          int kk = e & 31, n = e >> 5;
          int gn = nBase + n, gk = k0 + kk;
          float v = (gn < N && gk < K) ? B[(long long)gn * ldb + gk] : 0.f;
          Bs[n * LDSS + kk] = f2bf(v);
        }
      }
    }
    __syncthreads();

    FragU af[4], bfr[4];
    // A frag per ISA layout: lane row = l; K = {8*hi..8*hi+7} then {16+8*hi..16+8*hi+7}
#pragma unroll
    for (int mi = 0; mi < 4; ++mi) {
      const unsigned short* ap = &As[(wR + 16 * mi + l) * LDSS];
      af[mi].h[0] = *(const u16x8*)(ap + 8 * hi);
      af[mi].h[1] = *(const u16x8*)(ap + 16 + 8 * hi);
    }
    // B frag: lane col = l; K = {8*hi..8*hi+7} then {16+8*hi..16+8*hi+7}
#pragma unroll
    for (int ni = 0; ni < 4; ++ni) {
      const unsigned short* bp = &Bs[(wC + 16 * ni + l) * LDSS];
      bfr[ni].h[0] = *(const u16x8*)(bp + 8 * hi);
      bfr[ni].h[1] = *(const u16x8*)(bp + 16 + 8 * hi);
    }
#pragma unroll
    for (int mi = 0; mi < 4; ++mi)
#pragma unroll
      for (int ni = 0; ni < 4; ++ni)
        acc[mi][ni] = __builtin_amdgcn_wmma_f32_16x16x32_bf16(
            false, af[mi].v, false, bfr[ni].v, (short)0, acc[mi][ni], false, false);
  }

  // epilogue: D layout lane l -> col, VGPR v -> row v + 8*hi
#pragma unroll
  for (int mi = 0; mi < 4; ++mi) {
#pragma unroll
    for (int ni = 0; ni < 4; ++ni) {
      int col = nBase + wC + 16 * ni + l;
      if (col >= N) continue;
      float bv = (flags & 2) ? bias[col] : 0.f;
#pragma unroll
      for (int v = 0; v < 8; ++v) {
        int row = mBase + wR + 16 * mi + v + 8 * hi;
        if (row < M) C[(long long)row * ldc + col] = acc[mi][ni][v] + bv;
      }
    }
  }
}

// ---------------- elementwise / reduction kernels ----------------
__global__ void __launch_bounds__(256)
ln_kernel(const float* __restrict__ x, const float* __restrict__ g,
          const float* __restrict__ b, float* __restrict__ y, int D)
{
  int row = blockIdx.x;
  const float* xr = x + (long long)row * D;
  float* yr = y + (long long)row * D;
  __shared__ float red[256];
  int tid = threadIdx.x;
  float v[4]; int cnt = 0; float s = 0.f;
  for (int i = tid; i < D; i += 256) { float t = xr[i]; v[cnt++] = t; s += t; }
  red[tid] = s; __syncthreads();
  for (int o = 128; o; o >>= 1) { if (tid < o) red[tid] += red[tid + o]; __syncthreads(); }
  float mean = red[0] / D; __syncthreads();
  float s2 = 0.f;
  for (int i = 0; i < cnt; ++i) { float d = v[i] - mean; s2 += d * d; }
  red[tid] = s2; __syncthreads();
  for (int o = 128; o; o >>= 1) { if (tid < o) red[tid] += red[tid + o]; __syncthreads(); }
  float inv = rsqrtf(red[0] / D + EPS);
  cnt = 0;
  for (int i = tid; i < D; i += 256) { yr[i] = (v[cnt++] - mean) * inv * g[i] + b[i]; }
}

__global__ void __launch_bounds__(256)
softmax_kernel(float* __restrict__ x, int cols)
{
  long long row = blockIdx.x;
  float* r = x + row * (long long)cols;
  __shared__ float red[256];
  int tid = threadIdx.x;
  float v[4]; int cnt = 0; float mx = -3.4e38f;
  for (int i = tid; i < cols; i += 256) { float t = r[i]; v[cnt++] = t; mx = fmaxf(mx, t); }
  red[tid] = mx; __syncthreads();
  for (int o = 128; o; o >>= 1) { if (tid < o) red[tid] = fmaxf(red[tid], red[tid + o]); __syncthreads(); }
  mx = red[0]; __syncthreads();
  float s = 0.f;
  for (int i = 0; i < cnt; ++i) { v[i] = expf(v[i] - mx); s += v[i]; }
  red[tid] = s; __syncthreads();
  for (int o = 128; o; o >>= 1) { if (tid < o) red[tid] += red[tid + o]; __syncthreads(); }
  float inv = 1.f / red[0];
  cnt = 0;
  for (int i = tid; i < cols; i += 256) r[i] = v[cnt++] * inv;
}

__global__ void swish_kernel(float* x, int n) {
  int i = blockIdx.x * 256 + threadIdx.x; if (i >= n) return;
  float v = x[i]; x[i] = v / (1.f + expf(-v));
}

__global__ void combine_kernel(float* x, const float* f, float s, int n) {
  int i = blockIdx.x * 256 + threadIdx.x; if (i >= n) return;
  x[i] = s * (x[i] + f[i]);
}

__global__ void biasadd_hk_kernel(const float* q, const float* bias, float* out, int n) {
  int i = blockIdx.x * 256 + threadIdx.x; if (i >= n) return;
  int h = i / (TSEQ * DKH);
  int k = i % DKH;
  out[i] = q[i] + bias[h * DKH + k];
}

// scores[t,s] += bd[t, s + T-1 - t]   (Transformer-XL rel_shift)
__global__ void shiftadd_kernel(float* sc, const float* bd, int n) {
  int i = blockIdx.x * 256 + threadIdx.x; if (i >= n) return;
  int t = i / TSEQ, s2 = i % TSEQ;
  sc[i] += bd[(long long)t * PLEN + s2 + TSEQ - 1 - t];
}

__global__ void glu_kernel(const float* c, float* o, int n) {
  int i = blockIdx.x * 256 + threadIdx.x; if (i >= n) return;
  int t = i / DCONV, j = i % DCONV;
  float a = c[(long long)t * DFF + j];
  float g = c[(long long)t * DFF + DCONV + j];
  o[i] = a / (1.f + expf(-g));
}

__global__ void dwconv_kernel(const float* c, const float* w, float* o, int n) {
  int i = blockIdx.x * 256 + threadIdx.x; if (i >= n) return;
  int t = i / DCONV, ch = i % DCONV;
  float s = 0.f;
  for (int kk = 0; kk < KC; ++kk) {
    int tt = t + kk - 16;
    if (tt >= 0 && tt < TSEQ) s += c[(long long)tt * DCONV + ch] * w[ch * KC + kk];
  }
  o[i] = s;
}

__global__ void posemb_kernel(float* pe, int n) {
  int i = blockIdx.x * 256 + threadIdx.x; if (i >= n) return;
  int j = i / DM, d = i % DM;
  float div = expf(-(float)(d & ~1) * 9.210340371976184f / 512.f);  // ln(1e4)
  float ang = (float)(1022 - j) * div;
  pe[i] = (d & 1) ? cosf(ang) : sinf(ang);
}

// front-end conv subsampling
__global__ void conv1_kernel(const float* in, const float* w, const float* bias, float* out) {
  int i = blockIdx.x * 256 + threadIdx.x;
  if (i >= 32 * 2047 * 39) return;
  int j = i % 39, t = (i / 39) % 2047, c = i / (39 * 2047);
  float s = bias[c];
  for (int di = 0; di < 3; ++di) {
    int y = 2 * t + di;
    for (int dj = 0; dj < 3; ++dj) {
      int x = 2 * j + dj;
      float v = (y < 4090) ? in[y * 80 + x] : 0.f;   // rows >= 4090 are zero-pad
      s += v * w[c * 9 + di * 3 + dj];
    }
  }
  out[i] = fmaxf(s, 0.f);
}

__global__ void conv2_kernel(const float* in, const float* w, const float* bias, float* out) {
  int i = blockIdx.x * 256 + threadIdx.x;
  if (i >= 32 * 1023 * 19) return;
  int j = i % 19, t = (i / 19) % 1023, c = i / (19 * 1023);
  float s = bias[c];
  for (int ci = 0; ci < 32; ++ci)
    for (int di = 0; di < 3; ++di)
      for (int dj = 0; dj < 3; ++dj)
        s += in[(ci * 2047 + 2 * t + di) * 39 + 2 * j + dj] * w[((c * 32 + ci) * 3 + di) * 3 + dj];
  out[i] = fmaxf(s, 0.f);
}

__global__ void feat_kernel(const float* c2o, float* feat) {
  int i = blockIdx.x * 256 + threadIdx.x;
  if (i >= TSEQ * 608) return;
  int f = i % 608, t = i / 608;
  int c = f / 19, j = f % 19;
  feat[i] = c2o[(c * 1023 + t) * 19 + j];
}

// ---------------- host helpers ----------------
static inline void launch_gemm(hipStream_t st, const float* A, const float* B, const float* bias,
                               float* C, int M, int N, int K, int lda, int ldb, int ldc,
                               long long sA, long long sB, long long sC, int batch, int flags) {
  dim3 grid((N + 127) / 128, (M + 127) / 128, batch);
  gemm_bf16_kernel<<<grid, 128, 0, st>>>(A, B, bias, C, M, N, K, lda, ldb, ldc, sA, sB, sC, flags);
}
static inline dim3 ew(int n) { return dim3((n + 255) / 256); }

extern "C" void kernel_launch(void* const* d_in, const int* in_sizes, int n_in,
                              void* d_out, int out_size, void* d_ws, size_t ws_size,
                              hipStream_t stream) {
  (void)in_sizes; (void)n_in; (void)out_size; (void)ws_size;
  auto IN = [&](int i) { return (const float*)d_in[i]; };

  // ---- workspace layout (floats) ----
  float* W = (float*)d_ws;
  size_t off = 0;
  auto alloc = [&](size_t n) { float* p = W + off; off += n; return p; };
  float* xbuf   = alloc((size_t)TSEQ * DM);
  float* xn     = alloc((size_t)TSEQ * DM);
  float* tmp    = alloc((size_t)TSEQ * DM);
  float* hbuf   = alloc((size_t)TSEQ * DFF);
  float* qbuf   = alloc((size_t)TSEQ * DM);
  float* kbuf   = alloc((size_t)TSEQ * DM);
  float* vbuf   = alloc((size_t)TSEQ * DM);
  float* qu     = alloc((size_t)TSEQ * DM);
  float* qv     = alloc((size_t)TSEQ * DM);
  float* obuf   = alloc((size_t)TSEQ * DM);
  float* ppbuf  = alloc((size_t)NHEAD * PLEN * DKH);
  float* pe     = alloc((size_t)PLEN * DM);
  float* cbuf   = alloc((size_t)TSEQ * DFF);
  float* c2     = alloc((size_t)TSEQ * DCONV);
  float* c3     = alloc((size_t)TSEQ * DCONV);
  float* scores = alloc((size_t)NHEAD * TSEQ * TSEQ);
  float* bdws   = alloc((size_t)TSEQ * PLEN);
  // front-end temporaries reuse the scores region (it is only needed later)
  float* c1out = scores;            // 32*2047*39 = 2,554,656
  float* c2out = scores + 2600000;  // 32*1023*19 =   621,984
  float* feat  = scores + 3300000;  // 1023*608   =   621,984

  // ---- positional encoding (layer independent) ----
  posemb_kernel<<<ew(PLEN * DM), 256, 0, stream>>>(pe, PLEN * DM);

  // ---- conv subsampling front-end ----
  conv1_kernel<<<ew(32 * 2047 * 39), 256, 0, stream>>>(IN(0), IN(1), IN(2), c1out);
  conv2_kernel<<<ew(32 * 1023 * 19), 256, 0, stream>>>(c1out, IN(3), IN(4), c2out);
  feat_kernel<<<ew(TSEQ * 608), 256, 0, stream>>>(c2out, feat);
  // x = feat @ out_w + out_b   (K = 608 = 19*32)
  launch_gemm(stream, feat, IN(5), IN(6), xbuf, TSEQ, DM, 608, 608, DM, DM, 0, 0, 0, 1, 2);

  // ---- conformer layers ----
  const int NTD = TSEQ * DM;
  for (int L = 0; L < NL; ++L) {
    const float* f1_g  = IN(7)  + (long long)L * DM;
    const float* f1_b  = IN(8)  + (long long)L * DM;
    const float* f1_w1 = IN(9)  + (long long)L * DM * DFF;
    const float* f1_b1 = IN(10) + (long long)L * DFF;
    const float* f1_w2 = IN(11) + (long long)L * DFF * DM;
    const float* f1_b2 = IN(12) + (long long)L * DM;
    const float* f2_g  = IN(13) + (long long)L * DM;
    const float* f2_b  = IN(14) + (long long)L * DM;
    const float* f2_w1 = IN(15) + (long long)L * DM * DFF;
    const float* f2_b1 = IN(16) + (long long)L * DFF;
    const float* f2_w2 = IN(17) + (long long)L * DFF * DM;
    const float* f2_b2 = IN(18) + (long long)L * DM;
    const float* lnq_g = IN(19) + (long long)L * DM;
    const float* lnq_b = IN(20) + (long long)L * DM;
    const float* lnk_g = IN(21) + (long long)L * DM;
    const float* lnk_b = IN(22) + (long long)L * DM;
    const float* lnv_g = IN(23) + (long long)L * DM;
    const float* lnv_b = IN(24) + (long long)L * DM;
    const float* wq    = IN(25) + (long long)L * NHEAD * DM * DKH;
    const float* wk    = IN(26) + (long long)L * NHEAD * DM * DKH;
    const float* wv    = IN(27) + (long long)L * NHEAD * DM * DKH;
    const float* wp    = IN(28) + (long long)L * NHEAD * DM * DKH;
    const float* wo    = IN(29) + (long long)L * NHEAD * DKH * DM;  // flattened (512,512)
    const float* bu    = IN(30) + (long long)L * NHEAD * DKH;
    const float* bv    = IN(31) + (long long)L * NHEAD * DKH;
    const float* cln_g = IN(32) + (long long)L * DM;
    const float* cln_b = IN(33) + (long long)L * DM;
    const float* pw1   = IN(34) + (long long)L * DM * DFF;
    const float* dw    = IN(35) + (long long)L * DCONV * KC;
    const float* bn_g  = IN(36) + (long long)L * DCONV;
    const float* bn_b  = IN(37) + (long long)L * DCONV;
    const float* pw2   = IN(38) + (long long)L * DCONV * DM;
    const float* blk_g = IN(39) + (long long)L * DM;
    const float* blk_b = IN(40) + (long long)L * DM;

    // ---- FFN1 (macaron): x = 0.5*(x + ffn(x)) ----
    ln_kernel<<<TSEQ, 256, 0, stream>>>(xbuf, f1_g, f1_b, xn, DM);
    launch_gemm(stream, xn, f1_w1, f1_b1, hbuf, TSEQ, DFF, DM, DM, DFF, DFF, 0, 0, 0, 1, 2);
    swish_kernel<<<ew(TSEQ * DFF), 256, 0, stream>>>(hbuf, TSEQ * DFF);
    launch_gemm(stream, hbuf, f1_w2, f1_b2, tmp, TSEQ, DM, DFF, DFF, DM, DM, 0, 0, 0, 1, 2);
    combine_kernel<<<ew(NTD), 256, 0, stream>>>(xbuf, tmp, 0.5f, NTD);

    // ---- rel-pos MHSA ----
    ln_kernel<<<TSEQ, 256, 0, stream>>>(xbuf, lnq_g, lnq_b, xn, DM);
    launch_gemm(stream, xn, wq, nullptr, qbuf, TSEQ, DKH, DM, DM, DKH, DKH,
                0, (long long)DM * DKH, (long long)TSEQ * DKH, NHEAD, 0);
    ln_kernel<<<TSEQ, 256, 0, stream>>>(xbuf, lnk_g, lnk_b, xn, DM);
    launch_gemm(stream, xn, wk, nullptr, kbuf, TSEQ, DKH, DM, DM, DKH, DKH,
                0, (long long)DM * DKH, (long long)TSEQ * DKH, NHEAD, 0);
    ln_kernel<<<TSEQ, 256, 0, stream>>>(xbuf, lnv_g, lnv_b, xn, DM);
    launch_gemm(stream, xn, wv, nullptr, vbuf, TSEQ, DKH, DM, DM, DKH, DKH,
                0, (long long)DM * DKH, (long long)TSEQ * DKH, NHEAD, 0);
    // pp[h] = pos_emb @ wp[h]
    launch_gemm(stream, pe, wp, nullptr, ppbuf, PLEN, DKH, DM, DM, DKH, DKH,
                0, (long long)DM * DKH, (long long)PLEN * DKH, NHEAD, 0);
    biasadd_hk_kernel<<<ew(NTD), 256, 0, stream>>>(qbuf, bu, qu, NTD);
    biasadd_hk_kernel<<<ew(NTD), 256, 0, stream>>>(qbuf, bv, qv, NTD);
    // ac: scores[h] = (q+bu)[h] @ k[h]^T      (batched over heads, transB)
    launch_gemm(stream, qu, kbuf, nullptr, scores, TSEQ, TSEQ, DKH, DKH, DKH, TSEQ,
                (long long)TSEQ * DKH, (long long)TSEQ * DKH, (long long)TSEQ * TSEQ, NHEAD, 1);
    // bd per head (keeps workspace small), then fused rel_shift add
    for (int h = 0; h < NHEAD; ++h) {
      launch_gemm(stream, qv + (long long)h * TSEQ * DKH, ppbuf + (long long)h * PLEN * DKH,
                  nullptr, bdws, TSEQ, PLEN, DKH, DKH, DKH, PLEN, 0, 0, 0, 1, 1);
      shiftadd_kernel<<<ew(TSEQ * TSEQ), 256, 0, stream>>>(
          scores + (long long)h * TSEQ * TSEQ, bdws, TSEQ * TSEQ);
    }
    softmax_kernel<<<NHEAD * TSEQ, 256, 0, stream>>>(scores, TSEQ);
    // o[t, h*64+k] = attn[h] @ v[h]   (K = T = 1023, zero-padded tail)
    launch_gemm(stream, scores, vbuf, nullptr, obuf, TSEQ, DKH, TSEQ, TSEQ, DKH, DM,
                (long long)TSEQ * TSEQ, (long long)TSEQ * DKH, (long long)DKH, NHEAD, 0);
    launch_gemm(stream, obuf, wo, nullptr, tmp, TSEQ, DM, DM, DM, DM, DM, 0, 0, 0, 1, 0);
    combine_kernel<<<ew(NTD), 256, 0, stream>>>(xbuf, tmp, 1.0f, NTD);

    // ---- convolution module ----
    ln_kernel<<<TSEQ, 256, 0, stream>>>(xbuf, cln_g, cln_b, xn, DM);
    launch_gemm(stream, xn, pw1, nullptr, cbuf, TSEQ, DFF, DM, DM, DFF, DFF, 0, 0, 0, 1, 0);
    glu_kernel<<<ew(TSEQ * DCONV), 256, 0, stream>>>(cbuf, c2, TSEQ * DCONV);
    dwconv_kernel<<<ew(TSEQ * DCONV), 256, 0, stream>>>(c2, dw, c3, TSEQ * DCONV);
    ln_kernel<<<TSEQ, 256, 0, stream>>>(c3, bn_g, bn_b, c3, DCONV);   // in-place (register cached)
    swish_kernel<<<ew(TSEQ * DCONV), 256, 0, stream>>>(c3, TSEQ * DCONV);
    launch_gemm(stream, c3, pw2, nullptr, tmp, TSEQ, DM, DCONV, DCONV, DM, DM, 0, 0, 0, 1, 0);
    combine_kernel<<<ew(NTD), 256, 0, stream>>>(xbuf, tmp, 1.0f, NTD);

    // ---- FFN2 (macaron) ----
    ln_kernel<<<TSEQ, 256, 0, stream>>>(xbuf, f2_g, f2_b, xn, DM);
    launch_gemm(stream, xn, f2_w1, f2_b1, hbuf, TSEQ, DFF, DM, DM, DFF, DFF, 0, 0, 0, 1, 2);
    swish_kernel<<<ew(TSEQ * DFF), 256, 0, stream>>>(hbuf, TSEQ * DFF);
    launch_gemm(stream, hbuf, f2_w2, f2_b2, tmp, TSEQ, DM, DFF, DFF, DM, DM, 0, 0, 0, 1, 2);
    combine_kernel<<<ew(NTD), 256, 0, stream>>>(xbuf, tmp, 0.5f, NTD);

    // ---- final block LayerNorm (in-place safe) ----
    ln_kernel<<<TSEQ, 256, 0, stream>>>(xbuf, blk_g, blk_b, xbuf, DM);
  }

  hipMemcpyAsync(d_out, xbuf, sizeof(float) * (size_t)TSEQ * DM,
                 hipMemcpyDeviceToDevice, stream);
}
